// StreamingMultiheadAttention_73435350827141
// MI455X (gfx1250) — matrix operand order repaired and checked by reading, not compiled, not run on hardware
//
#include <hip/hip_runtime.h>
#include <hip/hip_bf16.h>

// ---------------------------------------------------------------------------
// Types for CDNA5 WMMA (wave32): 16x16x32 f16 -> f32 accumulate
// ---------------------------------------------------------------------------
typedef __attribute__((ext_vector_type(16))) _Float16 v16h;
typedef __attribute__((ext_vector_type(8)))  _Float16 v8h;
typedef __attribute__((ext_vector_type(8)))  float    v8f;

union FragU {
    v16h v;
    struct { v8h lo, hi; } h;
};

// A-operand fragment (16x32, MxK), ISA 7.12.2 layout:
//   lane L: row m = L&15 ; holds K = kb..kb+7 and kb+16..kb+23, kb = (L>=16)*8
__device__ __forceinline__ v16h load_a_frag(const _Float16* __restrict__ A,
                                            int lda, int m0, int k0, int lane) {
    const int row = m0 + (lane & 15);
    const int kb  = k0 + ((lane >> 4) << 3);      // 0 or 8
    const _Float16* p = A + (size_t)row * lda + kb;
    FragU u;
    u.h.lo = *(const v8h*)(p);                    // K kb .. kb+7
    u.h.hi = *(const v8h*)(p + 16);               // K kb+16 .. kb+23
    return u.v;
}

// B-operand fragment (32x16, KxN) from NT memory B[N][K]:
//   lane L: col n = L&15 ; holds K = kb..kb+7 and kb+16..kb+23, kb = (L>=16)*8
__device__ __forceinline__ v16h load_b_frag(const _Float16* __restrict__ B,
                                            int ldb, int n0, int k0, int lane) {
    const int col = n0 + (lane & 15);
    const int kb  = k0 + ((lane >> 4) << 3);      // 0 or 8
    const _Float16* p = B + (size_t)col * ldb + kb;
    FragU u;
    u.h.lo = *(const v8h*)(p);                    // K kb .. kb+7
    u.h.hi = *(const v8h*)(p + 16);               // K kb+16 .. kb+23
    return u.v;
}

__device__ __forceinline__ v8f wmma_f16(v16h a, v16h b, v8f c) {
    return __builtin_amdgcn_wmma_f32_16x16x32_f16(
        /*neg_a=*/false, a, /*neg_b=*/false, b,
        /*c_mod=*/(short)0, c, /*reuse_a=*/false, /*reuse_b=*/false);
}

// ---------------------------------------------------------------------------
// NT GEMM: C[M,N] = A[M,K] * B[N,K]^T, A/B f16 row-major, C f32 or f16.
// Block = 256 threads (8 waves, 2(M) x 4(N)). Wave tile 64x64 (4x4 frags):
// 16 b128 loads feed 16 WMMAs per K=32 step (1:1 load:wmma ratio).
// Block tile: 128 (M) x 256 (N). Requires M%128==0, N%256==0, K%32==0.
//
// flags bit0 (FLAG_SKIP_UPPER): skip output blocks entirely above the
//   diagonal (used for S = k@q^T: softmax row i never reads cols j > i).
// flags bit1 (FLAG_CAUSAL_K):   clamp K to (blockIdx.y+1)*128 (used for
//   P@V: P[i,j]==0 for j>i, so k >= rowblock_end contributes nothing).
// ---------------------------------------------------------------------------
#define FLAG_SKIP_UPPER 1
#define FLAG_CAUSAL_K   2

template <bool OUT_F16>
__global__ __launch_bounds__(256)
void gemm_nt_wmma(const _Float16* __restrict__ A,
                  const _Float16* __restrict__ B,
                  void* __restrict__ Cp,
                  int M, int N, int K, int lda, int ldb, int ldc, int flags) {
    const int mB = blockIdx.y * 128;
    const int nB = blockIdx.x * 256;
    if ((flags & FLAG_SKIP_UPPER) && (nB > mB + 127)) return;  // uniform exit

    int Keff = K;
    if (flags & FLAG_CAUSAL_K) {
        const int lim = mB + 128;
        Keff = (lim < K) ? lim : K;
    }

    const int lane = threadIdx.x & 31;
    const int wave = threadIdx.x >> 5;           // 0..7
    const int wr   = wave >> 2;                  // 0..1  (M direction)
    const int wc   = wave & 3;                   // 0..3  (N direction)
    const int m0   = mB + wr * 64;
    const int n0   = nB + wc * 64;

    v8f acc[4][4] = {};

    for (int k0 = 0; k0 < Keff; k0 += 32) {
        v16h af[4], bf[4];
        #pragma unroll
        for (int i = 0; i < 4; ++i)
            af[i] = load_a_frag(A, lda, m0 + 16 * i, k0, lane);
        #pragma unroll
        for (int j = 0; j < 4; ++j)
            bf[j] = load_b_frag(B, ldb, n0 + 16 * j, k0, lane);
        #pragma unroll
        for (int i = 0; i < 4; ++i)
            #pragma unroll
            for (int j = 0; j < 4; ++j)
                acc[i][j] = wmma_f16(af[i], bf[j], acc[i][j]);
    }

    // C/D layout: lane L -> col n = L&15 ; VGPR r -> row ((L>=16)?8:0)+r
    const int cn = lane & 15;
    const int mb = (lane >> 4) << 3;             // 0 or 8
    #pragma unroll
    for (int i = 0; i < 4; ++i) {
        #pragma unroll
        for (int j = 0; j < 4; ++j) {
            const int tm = m0 + 16 * i;
            const int tn = n0 + 16 * j;
            v8f a = acc[i][j];
            #pragma unroll
            for (int r = 0; r < 8; ++r) {
                const size_t idx = (size_t)(tm + mb + r) * ldc + (tn + cn);
                if (OUT_F16) ((_Float16*)Cp)[idx] = (_Float16)a[r];
                else         ((float*)Cp)[idx]    = a[r];
            }
        }
    }
}

// ---------------------------------------------------------------------------
// f32 -> f16 convert (grid-stride)
// ---------------------------------------------------------------------------
__global__ void f32_to_f16_kernel(const float* __restrict__ in,
                                  _Float16* __restrict__ out, long n) {
    long i = (long)blockIdx.x * blockDim.x + threadIdx.x;
    long stride = (long)gridDim.x * blockDim.x;
    for (; i < n; i += stride) out[i] = (_Float16)in[i];
}

// ---------------------------------------------------------------------------
// f16 transpose: VT[cols][rows] = V[rows][cols]; 32x32 LDS tile, block 32x8
// ---------------------------------------------------------------------------
__global__ __launch_bounds__(256)
void transpose_f16_kernel(const _Float16* __restrict__ V, int ldv,
                          _Float16* __restrict__ VT, int ldo) {
    __shared__ _Float16 tile[32][33];
    const int c0 = blockIdx.x * 32;
    const int r0 = blockIdx.y * 32;
    const int tx = threadIdx.x;                  // 0..31
    const int ty = threadIdx.y;                  // 0..7
    #pragma unroll
    for (int y = ty; y < 32; y += 8)
        tile[y][tx] = V[(size_t)(r0 + y) * ldv + (c0 + tx)];
    __syncthreads();
    #pragma unroll
    for (int y = ty; y < 32; y += 8)
        VT[(size_t)(c0 + y) * ldo + (r0 + tx)] = tile[tx][y];
}

// ---------------------------------------------------------------------------
// Causal softmax over row i of S[T,T]; valid columns j <= i; write f16 probs
// (zeros for masked columns — GEMM P@V depends on these exact zeros).
// One 256-thread block per row.
// ---------------------------------------------------------------------------
__global__ __launch_bounds__(256)
void softmax_causal_kernel(const float* __restrict__ S,
                           _Float16* __restrict__ P, int T) {
    const int i   = blockIdx.x;
    const int tid = threadIdx.x;
    const float* __restrict__ row = S + (size_t)i * T;
    _Float16* __restrict__ prow   = P + (size_t)i * T;
    const int n = i + 1;                         // valid length

    __shared__ float red[256];

    float m = -3.4e38f;
    for (int j = tid; j < n; j += 256) m = fmaxf(m, row[j]);
    red[tid] = m;
    __syncthreads();
    #pragma unroll
    for (int s = 128; s > 0; s >>= 1) {
        if (tid < s) red[tid] = fmaxf(red[tid], red[tid + s]);
        __syncthreads();
    }
    m = red[0];
    __syncthreads();

    float sum = 0.f;
    for (int j = tid; j < n; j += 256) sum += __expf(row[j] - m);
    red[tid] = sum;
    __syncthreads();
    #pragma unroll
    for (int s = 128; s > 0; s >>= 1) {
        if (tid < s) red[tid] += red[tid + s];
        __syncthreads();
    }
    sum = red[0];

    const float inv = 1.0f / sum;
    for (int j = tid; j < T; j += 256) {
        float pv = (j < n) ? __expf(row[j] - m) * inv : 0.0f;
        prow[j] = (_Float16)pv;
    }
}

// ---------------------------------------------------------------------------
// Host-side launch. Workspace layout (bytes, MiB-aligned carving):
//   xb    @   0  :  8 MiB  f16[4096,1024]
//   winb  @   8  :  6 MiB  f16[3072,1024]
//   woutb @  14  :  2 MiB  f16[1024,1024]
//   qkvb  @  16  : 24 MiB  f16[4096,3072]  (q|k|v along columns)
//   vtb   @  40  :  8 MiB  f16[1024,4096]  (v transposed)
//   aob   @  48  :  8 MiB  f16[4096,1024]
//   S     @  56  : 64 MiB  f32[4096,4096]  (only lower-triangular blocks written/read)
//   Pb    @ 120  : 32 MiB  f16[4096,4096]
// ---------------------------------------------------------------------------
extern "C" void kernel_launch(void* const* d_in, const int* in_sizes, int n_in,
                              void* d_out, int out_size, void* d_ws, size_t ws_size,
                              hipStream_t stream) {
    (void)in_sizes; (void)n_in; (void)out_size; (void)ws_size;
    const int T = 4096, D = 1024;

    const float* x     = (const float*)d_in[0];
    const float* w_in  = (const float*)d_in[1];
    const float* w_out = (const float*)d_in[2];
    float*       out   = (float*)d_out;

    char* ws = (char*)d_ws;
    _Float16* xb    = (_Float16*)(ws);
    _Float16* winb  = (_Float16*)(ws + ((size_t)8  << 20));
    _Float16* woutb = (_Float16*)(ws + ((size_t)14 << 20));
    _Float16* qkvb  = (_Float16*)(ws + ((size_t)16 << 20));
    _Float16* vtb   = (_Float16*)(ws + ((size_t)40 << 20));
    _Float16* aob   = (_Float16*)(ws + ((size_t)48 << 20));
    float*    S     = (float*)   (ws + ((size_t)56 << 20));
    _Float16* Pb    = (_Float16*)(ws + ((size_t)120 << 20));

    // 1) downconvert inputs to f16
    f32_to_f16_kernel<<<1024, 256, 0, stream>>>(x,     xb,    (long)T * D);
    f32_to_f16_kernel<<<1024, 256, 0, stream>>>(w_in,  winb,  (long)3 * D * D);
    f32_to_f16_kernel<<<1024, 256, 0, stream>>>(w_out, woutb, (long)D * D);

    // 2) QKV projection: qkv[T,3D] = x[T,D] @ w_in[3D,D]^T   (NT, f16 out)
    {
        dim3 grid(3 * D / 256, T / 128);
        gemm_nt_wmma<true><<<grid, 256, 0, stream>>>(
            xb, winb, qkvb, T, 3 * D, D, /*lda=*/D, /*ldb=*/D, /*ldc=*/3 * D, 0);
    }
    const _Float16* qb = qkvb;                 // cols [0,   D)
    const _Float16* kb = qkvb + D;             // cols [D,  2D)
    const _Float16* vb = qkvb + 2 * D;         // cols [2D, 3D)

    // 3) transpose v -> vtb[D,T] so P@v becomes an NT GEMM
    {
        dim3 grid(D / 32, T / 32), block(32, 8);
        transpose_f16_kernel<<<grid, block, 0, stream>>>(vb, 3 * D, vtb, T);
    }

    // 4) scores: S[T,T] = k @ q^T  (NT, f32 out); rows = key pos, cols = query
    //    pos. Skip blocks strictly above the diagonal (never read by softmax).
    {
        dim3 grid(T / 256, T / 128);
        gemm_nt_wmma<false><<<grid, 256, 0, stream>>>(
            kb, qb, S, T, T, D, /*lda=*/3 * D, /*ldb=*/3 * D, /*ldc=*/T,
            FLAG_SKIP_UPPER);
    }

    // 5) causal softmax rows of S -> Pb (f16, masked cols zeroed)
    softmax_causal_kernel<<<T, 256, 0, stream>>>(S, Pb, T);

    // 6) attention output: ao[T,D] = P[T,T] @ v[T,D] == P @ (vtb)^T  (NT, f16
    //    out). K clamped per row-block: P[i,k]==0 for k > i.
    {
        dim3 grid(D / 256, T / 128);
        gemm_nt_wmma<true><<<grid, 256, 0, stream>>>(
            Pb, vtb, aob, T, D, T, /*lda=*/T, /*ldb=*/T, /*ldc=*/D,
            FLAG_CAUSAL_K);
    }

    // 7) output projection: out[T,D] = ao @ w_out[D,D]^T  (NT, f32 out)
    {
        dim3 grid(D / 256, T / 128);
        gemm_nt_wmma<false><<<grid, 256, 0, stream>>>(
            aob, woutb, out, T, D, D, /*lda=*/D, /*ldb=*/D, /*ldc=*/D, 0);
    }
}
